// DistSAGE_5128190952006
// MI455X (gfx1250) — matrix operand order repaired and checked by reading, not compiled, not run on hardware
//
#include <hip/hip_runtime.h>

// ---------------- problem constants ----------------
#define NDST0 61952
#define NDST1 5632
#define NDST2 512
#define E0 619520
#define E1 56320
#define E2 5120
#define DIM 256
#define NC 19
#define NC_PAD 64

typedef __attribute__((ext_vector_type(16))) __bf16 v16bf;
typedef __attribute__((ext_vector_type(4)))  __bf16 v4bf;
typedef __attribute__((ext_vector_type(8)))  float  v8f;

// ---------------- zero agg + deg ----------------
__global__ void zero2_kernel(float* __restrict__ a, long na,
                             float* __restrict__ b, long nb) {
    long i = (long)blockIdx.x * blockDim.x + threadIdx.x;
    long stride = (long)gridDim.x * blockDim.x;
    for (long j = i; j < na; j += stride) a[j] = 0.0f;
    for (long j = i; j < nb; j += stride) b[j] = 0.0f;
}

// ---------------- f32 -> bf16 bulk convert (float4 / v4bf vectorized) -------
__global__ void tobf16_kernel(const float* __restrict__ src,
                              __bf16* __restrict__ dst, long n4) {
    long i = (long)blockIdx.x * blockDim.x + threadIdx.x;
    if (i >= n4) return;
    float4 v = ((const float4*)src)[i];
    v4bf t;
    t[0] = (__bf16)v.x; t[1] = (__bf16)v.y;
    t[2] = (__bf16)v.z; t[3] = (__bf16)v.w;
    ((v4bf*)dst)[i] = t;
}

// ---------------- W [K][N] f32 -> Wt [Npad][K] bf16 (transpose + pad) -------
__global__ void wtrans_kernel(const float* __restrict__ W,
                              __bf16* __restrict__ Wt, int N, int Npad) {
    int idx = blockIdx.x * blockDim.x + threadIdx.x;   // over Npad*256
    if (idx >= Npad * DIM) return;
    int n = idx / DIM, k = idx % DIM;                  // writes coalesced in k
    float v = (n < N) ? W[(long)k * N + n] : 0.0f;
    Wt[(long)n * DIM + k] = (__bf16)v;
}

// ---------------- edge-parallel scatter: one wave32 per edge ----------------
// Lane reads 2x float4 (b128, coalesced 1KB/row) then 8x global_atomic_add_f32.
__global__ void sage_scatter_kernel(const float* __restrict__ h,
                                    const int* __restrict__ esrc,
                                    const int* __restrict__ edst,
                                    float* __restrict__ agg,
                                    float* __restrict__ deg,
                                    int nedges) {
    int e    = (int)(((long)blockIdx.x * blockDim.x + threadIdx.x) >> 5);
    int lane = threadIdx.x & 31;
    if (e >= nedges) return;
    int s = esrc[e];
    int d = edst[e];
    const float4* src4 = (const float4*)(h + (long)s * DIM);
    float*        dst  = agg + (long)d * DIM;

    float4 v0 = src4[lane];
    float4 v1 = src4[lane + 32];

    int o = lane * 4;
    unsafeAtomicAdd(&dst[o + 0],   v0.x);
    unsafeAtomicAdd(&dst[o + 1],   v0.y);
    unsafeAtomicAdd(&dst[o + 2],   v0.z);
    unsafeAtomicAdd(&dst[o + 3],   v0.w);
    unsafeAtomicAdd(&dst[o + 128], v1.x);
    unsafeAtomicAdd(&dst[o + 129], v1.y);
    unsafeAtomicAdd(&dst[o + 130], v1.z);
    unsafeAtomicAdd(&dst[o + 131], v1.w);
    if (lane == 0) unsafeAtomicAdd(&deg[d], 1.0f);
}

// ---------------- agg/deg -> mean, emitted directly as bf16 ----------------
__global__ void sage_meanb_kernel(const float* __restrict__ agg,
                                  const float* __restrict__ deg,
                                  __bf16* __restrict__ meanb, int ndst) {
    long i = (long)blockIdx.x * blockDim.x + threadIdx.x;
    if (i >= (long)ndst * DIM) return;
    float dg = deg[i >> 8];
    meanb[i] = (__bf16)(agg[i] * (1.0f / fmaxf(dg, 1.0f)));
}

// ---------------- fused SAGE GEMM (all-bf16 inputs, async-LDS staged) -------
// Block tile 128x64, 8 wave32s, each wave a 32x32 region = 2x2 WMMA tiles.
// K loop: 16 tiles of 32 over K=512 (first 256 = Hdst@Wself, rest Mean@Wneigh).
// Staging = per-lane global_load_async_to_lds_b128, double-buffered, ASYNCcnt.
#define BM 128
#define BN 64
#define BK 32
#define ASTR 40   // bf16 elems; 80B row -> every 16B chunk stays 16B aligned
#define BSTR 40

__device__ __forceinline__ void issue_tile(const __bf16* __restrict__ Asrc,
                                           const __bf16* __restrict__ Wt,
                                           int brow, int bcol, int k0,
                                           __bf16* AsBuf, __bf16* BsBuf,
                                           int tid) {
    // A tile: 128 rows x 32 k = 512 x 16B chunks -> 2 per thread
#pragma unroll
    for (int j = 0; j < 2; ++j) {
        int f = tid + j * 256;
        int r = f >> 2;              // 4 chunks per 32-k row
        int c = (f & 3) * 8;         // bf16 element offset
        const void* g = (const void*)(Asrc + (long)(brow + r) * DIM + k0 + c);
        unsigned l = (unsigned)(uintptr_t)(AsBuf + r * ASTR + c);
        asm volatile("global_load_async_to_lds_b128 %0, %1, off"
                     :: "v"(l), "v"(g) : "memory");
    }
    // B tile: 64 n-rows x 32 k = 256 x 16B chunks -> 1 per thread
    {
        int n = tid >> 2;
        int c = (tid & 3) * 8;
        const void* g = (const void*)(Wt + (long)(bcol + n) * DIM + k0 + c);
        unsigned l = (unsigned)(uintptr_t)(BsBuf + n * BSTR + c);
        asm volatile("global_load_async_to_lds_b128 %0, %1, off"
                     :: "v"(l), "v"(g) : "memory");
    }
}

__global__ __launch_bounds__(256) void sage_gemm_kernel(
    const __bf16* __restrict__ Ab,   const __bf16* __restrict__ Meanb,
    const __bf16* __restrict__ Wst,  const __bf16* __restrict__ Wnt,
    const float* __restrict__ bias,  float* __restrict__ Out,
    __bf16* __restrict__ Outb, int M, int N, int do_relu)
{
    __shared__ __bf16 As[2][BM * ASTR];   // [row][k]
    __shared__ __bf16 Bs[2][BN * BSTR];   // [col][k] (weights pre-transposed)

    const int tid  = threadIdx.x;
    const int lane = tid & 31;
    const int wave = tid >> 5;
    const int wrow = (wave >> 1) * 32;
    const int wcol = (wave & 1) * 32;
    const int brow = blockIdx.x * BM;
    const int bcol = blockIdx.y * BN;

    v8f acc[2][2] = {};

    // Prologue: stage tile 0 into buffer 0.
    issue_tile(Ab, Wst, brow, bcol, 0, As[0], Bs[0], tid);

    for (int t = 0; t < 16; ++t) {
        const int cur = t & 1;
        if (t + 1 < 16) {
            const int kt = (t + 1) * BK;
            const __bf16* Asrc = (kt < DIM) ? Ab  : Meanb;
            const __bf16* Wt   = (kt < DIM) ? Wst : Wnt;
            issue_tile(Asrc, Wt, brow, bcol, kt & (DIM - 1),
                       As[1 - cur], Bs[1 - cur], tid);
            // 3 newest outstanding = tile t+1; in-order => tile t complete.
            asm volatile("s_wait_asynccnt 0x3" ::: "memory");
        } else {
            asm volatile("s_wait_asynccnt 0x0" ::: "memory");
        }
        __syncthreads();   // tile t resident in LDS for all waves

        // Fragments per ISA VGPR layouts (wave32).
        // A (16x32): lane half -> K {kb..kb+7, 16+kb..16+kb+7}, kb=(lane>>4)*8
        // B (32x16): lane half -> K {kb..kb+7, 16+kb..16+kb+7}, same as A
        const int fr   = lane & 15;
        const int kb8  = (lane >> 4) * 8;
        const int kb16 = (lane >> 4) * 16;
        const __bf16* Ac = As[cur];
        const __bf16* Bc = Bs[cur];

        v16bf a[2], b[2];
#pragma unroll
        for (int mb = 0; mb < 2; ++mb) {
            const __bf16* ap = &Ac[(wrow + mb * 16 + fr) * ASTR];
#pragma unroll
            for (int j = 0; j < 8; ++j) {
                a[mb][j]     = ap[kb8 + j];
                a[mb][8 + j] = ap[16 + kb8 + j];
            }
        }
#pragma unroll
        for (int nb = 0; nb < 2; ++nb) {
            const __bf16* bp = &Bc[(wcol + nb * 16 + fr) * BSTR];
#pragma unroll
            for (int j = 0; j < 8; ++j) { b[nb][j] = bp[kb8 + j]; b[nb][8 + j] = bp[16 + kb8 + j]; }
        }

#pragma unroll
        for (int mb = 0; mb < 2; ++mb)
#pragma unroll
            for (int nb = 0; nb < 2; ++nb)
                acc[mb][nb] = __builtin_amdgcn_wmma_f32_16x16x32_bf16(
                    false, a[mb], false, b[nb], (short)0, acc[mb][nb],
                    false, false);

        __syncthreads();   // everyone done reading buf[cur] before overwrite
    }

    // Epilogue: C/D layout -> lane col = lane&15, row = i + (lane>>4)*8
    const int colL  = lane & 15;
    const int rbase = (lane >> 4) * 8;
#pragma unroll
    for (int mb = 0; mb < 2; ++mb) {
#pragma unroll
        for (int nb = 0; nb < 2; ++nb) {
            int gcol = bcol + wcol + nb * 16 + colL;
            if (gcol >= N) continue;
            float bv = bias[gcol];
#pragma unroll
            for (int i = 0; i < 8; ++i) {
                int grow = brow + wrow + mb * 16 + rbase + i;
                float v = acc[mb][nb][i] + bv;
                if (do_relu) v = fmaxf(v, 0.0f);
                Out[(long)grow * N + gcol] = v;
                if (Outb) Outb[(long)grow * DIM + gcol] = (__bf16)v;
            }
        }
    }
}

// ---------------- launcher ----------------
extern "C" void kernel_launch(void* const* d_in, const int* in_sizes, int n_in,
                              void* d_out, int out_size, void* d_ws, size_t ws_size,
                              hipStream_t stream) {
    const float* x      = (const float*)d_in[0];
    const int*   esrc0  = (const int*)d_in[1];
    const int*   edst0  = (const int*)d_in[2];
    const int*   esrc1  = (const int*)d_in[3];
    const int*   edst1  = (const int*)d_in[4];
    const int*   esrc2  = (const int*)d_in[5];
    const int*   edst2  = (const int*)d_in[6];
    const float* Wself0 = (const float*)d_in[7];
    const float* Wnei0  = (const float*)d_in[8];
    const float* b0     = (const float*)d_in[9];
    const float* Wself1 = (const float*)d_in[10];
    const float* Wnei1  = (const float*)d_in[11];
    const float* b1     = (const float*)d_in[12];
    const float* Wself2 = (const float*)d_in[13];
    const float* Wnei2  = (const float*)d_in[14];
    const float* b2     = (const float*)d_in[15];

    // ---- workspace carve-up (256B aligned) ----
    char* p = (char*)d_ws;
    size_t off = 0;
    auto carve = [&](size_t bytes) -> char* {
        char* r = p + off;
        off = (off + bytes + 255) & ~(size_t)255;
        return r;
    };
    float*  h0    = (float*)carve((size_t)NDST0 * DIM * 4);
    float*  h1    = (float*)carve((size_t)NDST1 * DIM * 4);
    float*  agg   = (float*)carve((size_t)NDST0 * DIM * 4);
    float*  deg   = (float*)carve((size_t)NDST0 * 4);
    __bf16* xb    = (__bf16*)carve((size_t)NDST0 * DIM * 2);
    __bf16* h0b   = (__bf16*)carve((size_t)NDST0 * DIM * 2);  // epilogue writes all M rows
    __bf16* h1b   = (__bf16*)carve((size_t)NDST1 * DIM * 2);  // epilogue writes all M rows
    __bf16* meanb = (__bf16*)carve((size_t)NDST0 * DIM * 2);
    __bf16* Wst0  = (__bf16*)carve((size_t)DIM * DIM * 2);
    __bf16* Wnt0  = (__bf16*)carve((size_t)DIM * DIM * 2);
    __bf16* Wst1  = (__bf16*)carve((size_t)DIM * DIM * 2);
    __bf16* Wnt1  = (__bf16*)carve((size_t)DIM * DIM * 2);
    __bf16* Wst2  = (__bf16*)carve((size_t)NC_PAD * DIM * 2);
    __bf16* Wnt2  = (__bf16*)carve((size_t)NC_PAD * DIM * 2);

    // ---- one-time preps: bf16 copy of x's dst rows, bf16 transposed weights
    {
        long n4 = (long)NDST0 * DIM / 4;
        tobf16_kernel<<<(n4 + 255) / 256, 256, 0, stream>>>(x, xb, n4);
        wtrans_kernel<<<DIM * DIM / 256, 256, 0, stream>>>(Wself0, Wst0, DIM, DIM);
        wtrans_kernel<<<DIM * DIM / 256, 256, 0, stream>>>(Wnei0,  Wnt0, DIM, DIM);
        wtrans_kernel<<<DIM * DIM / 256, 256, 0, stream>>>(Wself1, Wst1, DIM, DIM);
        wtrans_kernel<<<DIM * DIM / 256, 256, 0, stream>>>(Wnei1,  Wnt1, DIM, DIM);
        wtrans_kernel<<<NC_PAD * DIM / 256, 256, 0, stream>>>(Wself2, Wst2, NC, NC_PAD);
        wtrans_kernel<<<NC_PAD * DIM / 256, 256, 0, stream>>>(Wnei2,  Wnt2, NC, NC_PAD);
    }

    // ---- Layer 0: x -> h0 (relu) ----
    zero2_kernel<<<2048, 256, 0, stream>>>(agg, (long)NDST0 * DIM, deg, NDST0);
    sage_scatter_kernel<<<E0 / 8, 256, 0, stream>>>(x, esrc0, edst0, agg, deg, E0);
    sage_meanb_kernel<<<NDST0, 256, 0, stream>>>(agg, deg, meanb, NDST0);
    sage_gemm_kernel<<<dim3(NDST0 / BM, DIM / BN), 256, 0, stream>>>(
        xb, meanb, Wst0, Wnt0, b0, h0, h0b, NDST0, DIM, 1);

    // ---- Layer 1: h0 -> h1 (relu) ----
    zero2_kernel<<<2048, 256, 0, stream>>>(agg, (long)NDST1 * DIM, deg, NDST1);
    sage_scatter_kernel<<<E1 / 8, 256, 0, stream>>>(h0, esrc1, edst1, agg, deg, E1);
    sage_meanb_kernel<<<NDST1, 256, 0, stream>>>(agg, deg, meanb, NDST1);
    sage_gemm_kernel<<<dim3(NDST1 / BM, DIM / BN), 256, 0, stream>>>(
        h0b, meanb, Wst1, Wnt1, b1, h1, h1b, NDST1, DIM, 1);

    // ---- Layer 2: h1 -> out (no relu, N=19) ----
    zero2_kernel<<<512, 256, 0, stream>>>(agg, (long)NDST2 * DIM, deg, NDST2);
    sage_scatter_kernel<<<E2 / 8, 256, 0, stream>>>(h1, esrc2, edst2, agg, deg, E2);
    sage_meanb_kernel<<<NDST2, 256, 0, stream>>>(agg, deg, meanb, NDST2);
    sage_gemm_kernel<<<dim3(NDST2 / BM, 1), 256, 0, stream>>>(
        h1b, meanb, Wst2, Wnt2, b2, (float*)d_out, (__bf16*)nullptr, NDST2, NC, 0);
}
